// LSTM_79259326480934
// MI455X (gfx1250) — compile-verified, matrix operand order checked by reading, not run on hardware
//
#include <hip/hip_runtime.h>
#include <hip/hip_bf16.h>

// ---------------------------------------------------------------------------
// 3-layer LSTM (B=256, T=2048, I=128, H=64) + FC for gfx1250 (CDNA5).
//   * Prologue kernel converts x to bf16 once (memory-bound, ~0.4 GB moved).
//   * Per layer: 16 WGs (16 batch rows each) x 16 waves (one 16-wide gate
//     column tile each). Weight B-fragments resident in VGPRs all scan long.
//   * Per timestep: ONE Tensor-Data-Mover DMA (tensor_load_to_lds) stages the
//     16 x IN_DIM bf16 input tile into a double-buffered LDS tile (issued one
//     step ahead, drained with s_wait_tensorcnt before the step barrier);
//     v_wmma_f32_16x16x32_bf16 computes gates; branch-free fast
//     sigmoid/tanh (v_rcp_f32 + v_exp_f32); c/h update via LDS exchange.
//   * Final FC = memory-bound GEMV.
// ---------------------------------------------------------------------------

typedef __bf16 v16bf __attribute__((ext_vector_type(16)));
typedef __bf16 v8bf  __attribute__((ext_vector_type(8)));
typedef __bf16 v4bf  __attribute__((ext_vector_type(4)));
typedef __bf16 v2bf  __attribute__((ext_vector_type(2)));
typedef float  v8f   __attribute__((ext_vector_type(8)));
typedef unsigned int uint4v __attribute__((ext_vector_type(4)));
typedef int    int4v __attribute__((ext_vector_type(4)));
typedef int    int8v __attribute__((ext_vector_type(8)));

#define LSTM_B 256
#define LSTM_T 2048
#define LSTM_I 128
#define LSTM_H 64
#define LSTM_G 256   // 4*H

// Low 32 bits of a generic pointer to LDS = LDS byte offset (ISA aperture rule).
__device__ inline unsigned lds_addr_of(const void* p) {
  return (unsigned)(unsigned long long)(size_t)p;
}

// One TDM DMA: copy a 2D tile (nrows x row_elems bf16 elements, row stride
// row_stride_elems) from global memory into LDS at lds_off.
// D# packing per CDNA5 ISA 8.3/8.4 (group0: count/lds/global/type,
// group1: data_size=2B, dims, strides). Groups 2/3 zero (2D tensor).
__device__ inline void tdm_load_tile(const void* gptr, unsigned lds_off,
                                     unsigned row_elems, unsigned nrows,
                                     unsigned long long row_stride_elems) {
  const unsigned long long ga = (unsigned long long)(size_t)gptr;
  uint4v g0;
  g0[0] = 1u;                                            // count=1 (valid)
  g0[1] = lds_off;                                       // lds_addr
  g0[2] = (unsigned)(ga & 0xffffffffull);                // global_addr[31:0]
  g0[3] = (unsigned)((ga >> 32) & 0x01ffffffull) | (2u << 30);  // addr[56:32], type=2
  int8v g1;
  g1[0] = (int)(1u << 16);                               // data_size=1 -> 2 bytes
  g1[1] = (int)((row_elems & 0xffffu) << 16);            // tensor_dim0[15:0]
  g1[2] = (int)(((row_elems >> 16) & 0xffffu) |
                ((nrows & 0xffffu) << 16));              // td0 hi16 | td1 lo16
  g1[3] = (int)(((nrows >> 16) & 0xffffu) |
                ((row_elems & 0xffffu) << 16));          // td1 hi16 | tile_dim0
  g1[4] = (int)(nrows & 0xffffu);                        // tile_dim1 (tile_dim2=0)
  g1[5] = (int)(row_stride_elems & 0xffffffffull);       // tensor_dim0_stride lo32
  g1[6] = (int)((row_stride_elems >> 32) & 0xffffull);   // stride hi16 (td1_stride=0)
  g1[7] = 0;
  const int4v gz = {0, 0, 0, 0};
#if __has_include(<hip/amd_detail/amd_gfx1250_TDM.h>)
  const int8v gz8 = {0, 0, 0, 0, 0, 0, 0, 0};
  __builtin_amdgcn_tensor_load_to_lds(g0, g1, gz, gz, gz8, 0);
#else
  __builtin_amdgcn_tensor_load_to_lds(g0, g1, gz, gz, 0);
#endif
}

// Fast branch-free activations (v_exp_f32 + v_rcp_f32; no IEEE div sequence).
__device__ inline float ftanh_fast(float x) {
  return 2.0f * __builtin_amdgcn_rcpf(1.0f + __expf(-2.0f * x)) - 1.0f;
}

// A-fragment (16x32 bf16, row-major LDS source). ISA 7.12.2 layout.
__device__ inline v16bf load_a_frag(const __bf16* __restrict__ buf, int stride,
                                    int lane, int kb) {
  const int M    = lane & 15;
  const int base = (lane >> 4) << 3;
  const int K0   = kb * 32 + base;
  const v8bf lo = *(const v8bf*)(buf + M * stride + K0);
  const v8bf hi = *(const v8bf*)(buf + M * stride + K0 + 16);
  v16bf a;
#pragma unroll
  for (int i = 0; i < 8; i++) { a[i] = lo[i]; a[i + 8] = hi[i]; }
  return a;
}

// B-fragment (32x16 bf16) of w^T where w is (G x ld) f32 row-major.
__device__ inline v16bf load_b_frag(const float* __restrict__ w, int ld,
                                    int row_n, int lane, int kb) {
  const int base = (lane >> 4) << 3;
  const float* p0 = w + (size_t)row_n * ld + kb * 32 + base;
  const float* p1 = p0 + 16;
  v16bf b;
#pragma unroll
  for (int i = 0; i < 8; i++) {
    b[i]     = (__bf16)p0[i];
    b[i + 8] = (__bf16)p1[i];
  }
  return b;
}

template <int IN_DIM>
__global__ __launch_bounds__(512) void lstm_scan_kernel(
    const __bf16* __restrict__ xin,
    const float* __restrict__ w_ih, const float* __restrict__ w_hh,
    const float* __restrict__ b_ih, const float* __restrict__ b_hh,
    __bf16* __restrict__ hout) {
  constexpr int KB_IN  = IN_DIM / 32;
  constexpr int KB_ALL = KB_IN + 2;
  __shared__ alignas(16) __bf16 xbuf[2][16 * IN_DIM];  // TDM double buffer
  __shared__ alignas(16) __bf16 hbuf[16 * LSTM_H];
  __shared__ alignas(16) float  gbuf[16 * LSTM_G];
  __shared__ alignas(16) float  cbuf[16 * LSTM_H];

  const int tid  = threadIdx.x;
  const int lane = tid & 31;
  const int wid  = tid >> 5;            // wave id -> gate column tile
  const int b0   = blockIdx.x * 16;     // batch-row tile base
  const int n0   = wid * 16;
  const int nlan = n0 + (lane & 15);

  // ---- resident bf16 weight fragments + bias -----------------------------
  v16bf bw[KB_ALL];
#pragma unroll
  for (int kb = 0; kb < KB_IN; kb++)
    bw[kb] = load_b_frag(w_ih, IN_DIM, nlan, lane, kb);
#pragma unroll
  for (int kb = 0; kb < 2; kb++)
    bw[KB_IN + kb] = load_b_frag(w_hh, LSTM_H, nlan, lane, kb);
  const float bias  = b_ih[nlan] + b_hh[nlan];
  const float mgate = ((wid >> 2) == 2) ? 2.0f : 1.0f;  // uniform: tanh gate

  for (int i = tid; i < 16 * LSTM_H; i += 512) {
    hbuf[i] = (__bf16)0.0f;
    cbuf[i] = 0.0f;
  }

  const __bf16* xtile0 = xin + (size_t)b0 * LSTM_T * IN_DIM;
  const unsigned long long row_stride = (unsigned long long)LSTM_T * IN_DIM;

  // prologue: DMA tile for t=0
  if (wid == 0) {
    tdm_load_tile(xtile0, lds_addr_of(&xbuf[0][0]), IN_DIM, 16, row_stride);
    __builtin_amdgcn_s_wait_tensorcnt(0);
  }
  __syncthreads();

  const int m_upd = tid >> 5;           // update phase: row
  const int j0    = (tid & 31) * 2;     // update phase: column pair
  const int mb    = (lane >> 4) << 3;   // C-layout row base per half-wave

  for (int t = 0; t < LSTM_T; t++) {
    const int cur = t & 1;
    // issue next step's DMA now; it lands while we compute this step
    if (wid == 0 && (t + 1) < LSTM_T) {
      tdm_load_tile(xtile0 + (size_t)(t + 1) * IN_DIM,
                    lds_addr_of(&xbuf[cur ^ 1][0]), IN_DIM, 16, row_stride);
    }

    // ---- batch all A-fragment LDS loads, then the WMMA chain -------------
    v16bf afr[KB_ALL];
#pragma unroll
    for (int kb = 0; kb < KB_IN; kb++)
      afr[kb] = load_a_frag(&xbuf[cur][0], IN_DIM, lane, kb);
#pragma unroll
    for (int kb = 0; kb < 2; kb++)
      afr[KB_IN + kb] = load_a_frag(hbuf, LSTM_H, lane, kb);

    v8f acc0, acc1;
#pragma unroll
    for (int v = 0; v < 8; v++) { acc0[v] = bias; acc1[v] = 0.0f; }
#pragma unroll
    for (int kb = 0; kb < KB_ALL; kb += 2)
      acc0 = __builtin_amdgcn_wmma_f32_16x16x32_bf16(
          false, afr[kb], false, bw[kb], (short)0, acc0, false, false);
#pragma unroll
    for (int kb = 1; kb < KB_ALL; kb += 2)
      acc1 = __builtin_amdgcn_wmma_f32_16x16x32_bf16(
          false, afr[kb], false, bw[kb], (short)0, acc1, false, false);

    // ---- branch-free activation: m*sigmoid(m*x) - (m-1)  (m=2 -> tanh) ---
#pragma unroll
    for (int v = 0; v < 8; v++) {
      const float xv = acc0[v] + acc1[v];
      const float s  = __builtin_amdgcn_rcpf(1.0f + __expf(-mgate * xv));
      gbuf[(mb + v) * LSTM_G + nlan] = mgate * s - (mgate - 1.0f);
    }
    __syncthreads();

    // ---- c/h update: one column pair per thread --------------------------
    {
      const int gi = m_upd * LSTM_G + j0;
      const int ci = m_upd * LSTM_H + j0;
      const float ig0 = gbuf[gi],       ig1 = gbuf[gi + 1];
      const float fg0 = gbuf[gi + 64],  fg1 = gbuf[gi + 65];
      const float gg0 = gbuf[gi + 128], gg1 = gbuf[gi + 129];
      const float og0 = gbuf[gi + 192], og1 = gbuf[gi + 193];
      const float c0 = fg0 * cbuf[ci]     + ig0 * gg0;
      const float c1 = fg1 * cbuf[ci + 1] + ig1 * gg1;
      cbuf[ci]     = c0;
      cbuf[ci + 1] = c1;
      v2bf hb;
      hb[0] = (__bf16)(og0 * ftanh_fast(c0));
      hb[1] = (__bf16)(og1 * ftanh_fast(c1));
      *(v2bf*)&hbuf[ci] = hb;
      *(v2bf*)(hout + ((size_t)(b0 + m_upd) * LSTM_T + t) * LSTM_H + j0) = hb;
    }
    if (wid == 0) __builtin_amdgcn_s_wait_tensorcnt(0);  // next tile landed
    __syncthreads();
  }
}

// Prologue: convert x (f32) to bf16 once, so the scan streams bf16 via TDM.
__global__ __launch_bounds__(256) void f32_to_bf16_kernel(
    const float* __restrict__ in, __bf16* __restrict__ out, size_t n4) {
  const size_t i = ((size_t)blockIdx.x * blockDim.x + threadIdx.x);
  if (i >= n4) return;
  const float4 f = ((const float4*)in)[i];
  v4bf o;
  o[0] = (__bf16)f.x; o[1] = (__bf16)f.y;
  o[2] = (__bf16)f.z; o[3] = (__bf16)f.w;
  ((v4bf*)out)[i] = o;
}

// Final FC: out[i] = dot(h[i,:], fc_w) + fc_b (memory-bound GEMV)
__global__ __launch_bounds__(256) void lstm_fc_kernel(
    const __bf16* __restrict__ h, const float* __restrict__ fc_w,
    const float* __restrict__ fc_b, float* __restrict__ out, int n) {
  const int i = blockIdx.x * blockDim.x + threadIdx.x;
  if (i >= n) return;
  const v8bf* row = (const v8bf*)(h + (size_t)i * LSTM_H);
  float s = fc_b[0];
#pragma unroll
  for (int k = 0; k < 8; k++) {
    const v8bf c = row[k];
#pragma unroll
    for (int j = 0; j < 8; j++) s += (float)c[j] * fc_w[k * 8 + j];
  }
  out[i] = s;
}

extern "C" void kernel_launch(void* const* d_in, const int* in_sizes, int n_in,
                              void* d_out, int out_size, void* d_ws,
                              size_t ws_size, hipStream_t stream) {
  const float* x     = (const float*)d_in[0];
  const float* w_ih0 = (const float*)d_in[1];
  const float* w_hh0 = (const float*)d_in[2];
  const float* b_ih0 = (const float*)d_in[3];
  const float* b_hh0 = (const float*)d_in[4];
  const float* w_ih1 = (const float*)d_in[5];
  const float* w_hh1 = (const float*)d_in[6];
  const float* b_ih1 = (const float*)d_in[7];
  const float* b_hh1 = (const float*)d_in[8];
  const float* w_ih2 = (const float*)d_in[9];
  const float* w_hh2 = (const float*)d_in[10];
  const float* b_ih2 = (const float*)d_in[11];
  const float* b_hh2 = (const float*)d_in[12];
  const float* fc_w  = (const float*)d_in[13];
  const float* fc_b  = (const float*)d_in[14];
  float* out = (float*)d_out;

  // workspace: xbf (128MB) + two ping-pong bf16 h buffers (64MB each)
  const size_t x_elems   = (size_t)LSTM_B * LSTM_T * LSTM_I;
  const size_t seq_elems = (size_t)LSTM_B * LSTM_T * LSTM_H;
  __bf16* xbf  = (__bf16*)d_ws;
  __bf16* buf0 = xbf + x_elems;
  __bf16* buf1 = buf0 + seq_elems;

  const size_t n4 = x_elems / 4;
  f32_to_bf16_kernel<<<(unsigned)((n4 + 255) / 256), 256, 0, stream>>>(x, xbf, n4);

  const dim3 grid(LSTM_B / 16), block(512);
  lstm_scan_kernel<LSTM_I><<<grid, block, 0, stream>>>(
      xbf, w_ih0, w_hh0, b_ih0, b_hh0, buf0);
  lstm_scan_kernel<LSTM_H><<<grid, block, 0, stream>>>(
      buf0, w_ih1, w_hh1, b_ih1, b_hh1, buf1);
  lstm_scan_kernel<LSTM_H><<<grid, block, 0, stream>>>(
      buf1, w_ih2, w_hh2, b_ih2, b_hh2, buf0);

  const int n = LSTM_B * LSTM_T;
  lstm_fc_kernel<<<(n + 255) / 256, 256, 0, stream>>>(buf0, fc_w, fc_b, out, n);
}
